// Net_72447508349482
// MI455X (gfx1250) — matrix operand order repaired and checked by reading, not compiled, not run on hardware
//
#include <hip/hip_runtime.h>
#include <math.h>

typedef __attribute__((ext_vector_type(16))) _Float16 v16h;
typedef __attribute__((ext_vector_type(8)))  _Float16 v8h;
typedef __attribute__((ext_vector_type(8)))  float    v8f;

#define BATCH   256
#define NC1     20
#define NC2     50
#define KSZ     5
#define FC1_IN  800
#define FC1_OUT 500
#define FC1_PAD 512
#define NCLS    10
#define NEG_INF (-3.402823466e38f)

// ---------------- conv1 (1->20, 5x5, valid) + 2x2 maxpool -> h1[256][20][12][12]
__global__ void conv1_pool_kernel(const float* __restrict__ x,
                                  const float* __restrict__ w,
                                  const float* __restrict__ bias,
                                  float* __restrict__ h1) {
    int idx = blockIdx.x * blockDim.x + threadIdx.x;
    if (idx >= BATCH * NC1 * 144) return;
    int px = idx % 12, py = (idx / 12) % 12;
    int c  = (idx / 144) % NC1;
    int b  = idx / (144 * NC1);
    const float* xb = x + (size_t)b * 28 * 28;
    float wr[25];
#pragma unroll
    for (int i = 0; i < 25; i++) wr[i] = w[c * 25 + i];
    float bc = bias[c];
    float m = NEG_INF;
#pragma unroll
    for (int dy = 0; dy < 2; dy++)
#pragma unroll
    for (int dx = 0; dx < 2; dx++) {
        int oy = 2 * py + dy, ox = 2 * px + dx;
        float s = bc;
#pragma unroll
        for (int kh = 0; kh < KSZ; kh++)
#pragma unroll
        for (int kw = 0; kw < KSZ; kw++)
            s += xb[(oy + kh) * 28 + ox + kw] * wr[kh * 5 + kw];
        m = fmaxf(m, s);
    }
    h1[idx] = m;
}

// ------- adder2d (L1 distance) + 2x2 maxpool + BN statistics. One block per batch.
__global__ void adder_pool_stats_kernel(const float* __restrict__ h1,
                                        const float* __restrict__ aw,
                                        float* __restrict__ h2,      // [256][800]
                                        float* __restrict__ bn_sums) // [sum(50), sumsq(50)]
{
    __shared__ float tile[NC1 * 144];   // h1[b] tile, 11.5 KB in LDS
    __shared__ float csum[NC2], csumsq[NC2];
    int b = blockIdx.x;
    for (int i = threadIdx.x; i < NC1 * 144; i += blockDim.x)
        tile[i] = h1[(size_t)b * NC1 * 144 + i];
    for (int i = threadIdx.x; i < NC2; i += blockDim.x) { csum[i] = 0.f; csumsq[i] = 0.f; }
    __syncthreads();

    for (int out = threadIdx.x; out < NC2 * 16; out += blockDim.x) {
        int p = out & 15, o = out >> 4;
        int py = p >> 2, px = p & 3;
        const float* wo = aw + o * 500;
        float m = NEG_INF;
#pragma unroll
        for (int dy = 0; dy < 2; dy++)
#pragma unroll
        for (int dx = 0; dx < 2; dx++) {
            int y = 2 * py + dy, xx = 2 * px + dx;
            float s = 0.f;
            for (int c = 0; c < NC1; c++) {
                const float* tc = tile + c * 144 + y * 12 + xx;
                const float* wc = wo + c * 25;
#pragma unroll
                for (int kh = 0; kh < KSZ; kh++)
#pragma unroll
                for (int kw = 0; kw < KSZ; kw++)
                    s += fabsf(tc[kh * 12 + kw] - wc[kh * 5 + kw]);
            }
            m = fmaxf(m, -s);
        }
        h2[(size_t)b * 800 + out] = m;
        atomicAdd(&csum[o], m);
        atomicAdd(&csumsq[o], m * m);
    }
    __syncthreads();
    for (int i = threadIdx.x; i < NC2; i += blockDim.x) {
        atomicAdd(&bn_sums[i], csum[i]);
        atomicAdd(&bn_sums[NC2 + i], csumsq[i]);
    }
}

// ---------------- BN finalize: per-channel scale/shift
__global__ void bn_finalize_kernel(const float* __restrict__ bn_sums,
                                   const float* __restrict__ gamma,
                                   const float* __restrict__ beta,
                                   float* __restrict__ bn_ab) {
    int c = threadIdx.x;
    if (c < NC2) {
        float n    = (float)(BATCH * 16);
        float mean = bn_sums[c] / n;
        float var  = bn_sums[NC2 + c] / n - mean * mean;
        float sc   = gamma[c] * rsqrtf(var + 1e-5f);
        bn_ab[c]       = sc;
        bn_ab[NC2 + c] = beta[c] - mean * sc;
    }
}

// ---------------- convert BN(h2) and fc1_w to f16 (w padded to 512 rows)
__global__ void to_f16_kernel(const float* __restrict__ h2,
                              const float* __restrict__ bn_ab,
                              const float* __restrict__ fc1_w,
                              _Float16* __restrict__ act,   // [256][800]
                              _Float16* __restrict__ w1) {  // [512][800]
    int idx = blockIdx.x * blockDim.x + threadIdx.x;
    const int nact = BATCH * FC1_IN;
    if (idx < nact) {
        int c = (idx % FC1_IN) >> 4;                 // j = ch*16 + pixel
        act[idx] = (_Float16)(h2[idx] * bn_ab[c] + bn_ab[NC2 + c]);
    } else {
        int k = idx - nact;
        if (k < FC1_PAD * FC1_IN) {
            int n = k / FC1_IN;
            w1[k] = (n < FC1_OUT) ? (_Float16)fc1_w[k] : (_Float16)0.f;
        }
    }
}

// ---------------- fc1 via WMMA f16 -> f32: out = relu(act @ w1^T + b). One wave/tile.
__global__ void fc1_wmma_kernel(const _Float16* __restrict__ act,
                                const _Float16* __restrict__ w1,
                                const float* __restrict__ fc1_b,
                                float* __restrict__ out) {   // [256][512] f32
    int wave = (blockIdx.x * blockDim.x + threadIdx.x) >> 5;
    int lane = threadIdx.x & 31;
    int nt   = wave % (FC1_PAD / 16);   // 0..31
    int mt   = wave / (FC1_PAD / 16);   // 0..15
    int half = lane >> 4;
    int l16  = lane & 15;

    const _Float16* pa = act + (size_t)(mt * 16 + l16) * FC1_IN;  // row M = l16
    const _Float16* pb = w1  + (size_t)(nt * 16 + l16) * FC1_IN;  // col N = l16 (B[K,N]=W[N,K])

    union U16 { v16h v; v8h h[2]; };
    v8f c = {};
#pragma unroll 5
    for (int k0 = 0; k0 < FC1_IN; k0 += 32) {
        U16 a, bm;
        // A 16x32 f16 layout: VGPR0-3 hold K = 8*half..8*half+7, VGPR4-7 hold K = 16+8*half..+7
        a.h[0]  = *(const v8h*)(pa + k0 + half * 8);
        a.h[1]  = *(const v8h*)(pa + k0 + 16 + half * 8);
        // B 32x16 f16 layout: VGPR0-3 hold K = 8*half..8*half+7, VGPR4-7 hold K = 16+8*half..+7 of column N
        bm.h[0] = *(const v8h*)(pb + k0 + half * 8);
        bm.h[1] = *(const v8h*)(pb + k0 + 16 + half * 8);
        c = __builtin_amdgcn_wmma_f32_16x16x32_f16(false, a.v, false, bm.v,
                                                   (short)0, c, false, false);
    }
    int ncol = nt * 16 + l16;
    float bias = (ncol < FC1_OUT) ? fc1_b[ncol] : 0.f;
#pragma unroll
    for (int v = 0; v < 8; v++) {
        int mrow = mt * 16 + half * 8 + v;   // C/D layout: VGPR v -> M = 8*half + v
        out[(size_t)mrow * FC1_PAD + ncol] = fmaxf(c[v] + bias, 0.f);
    }
}

// ---------------- fc2 (500->10) + log_softmax. One wave per batch row.
__global__ void fc2_lsm_kernel(const float* __restrict__ h,   // [256][512]
                               const float* __restrict__ w2,  // [10][500]
                               const float* __restrict__ b2,
                               float* __restrict__ out) {     // [256][10]
    int b = blockIdx.x;
    int lane = threadIdx.x;
    __shared__ float logits[NCLS];
    const float* hb = h + (size_t)b * FC1_PAD;
    for (int cls = 0; cls < NCLS; cls++) {
        float s = 0.f;
        for (int k = lane; k < FC1_OUT; k += 32)
            s += hb[k] * w2[cls * FC1_OUT + k];
#pragma unroll
        for (int off = 16; off >= 1; off >>= 1)
            s += __shfl_xor(s, off, 32);
        if (lane == 0) logits[cls] = s + b2[cls];
    }
    __syncthreads();
    float m = NEG_INF;
#pragma unroll
    for (int i = 0; i < NCLS; i++) m = fmaxf(m, logits[i]);
    float se = 0.f;
#pragma unroll
    for (int i = 0; i < NCLS; i++) se += expf(logits[i] - m);
    float lse = logf(se) + m;
    if (lane < NCLS) out[b * NCLS + lane] = logits[lane] - lse;
}

extern "C" void kernel_launch(void* const* d_in, const int* in_sizes, int n_in,
                              void* d_out, int out_size, void* d_ws, size_t ws_size,
                              hipStream_t stream) {
    const float* x       = (const float*)d_in[0];
    const float* conv1_w = (const float*)d_in[1];
    const float* conv1_b = (const float*)d_in[2];
    const float* adder_w = (const float*)d_in[3];
    const float* bn_g    = (const float*)d_in[4];
    const float* bn_b    = (const float*)d_in[5];
    const float* fc1_w   = (const float*)d_in[6];
    const float* fc1_b   = (const float*)d_in[7];
    const float* fc2_w   = (const float*)d_in[8];
    const float* fc2_b   = (const float*)d_in[9];

    char* ws = (char*)d_ws;
    size_t off = 0;
    auto alloc = [&](size_t bytes) -> void* {
        void* p = ws + off;
        off = (off + bytes + 255) & ~(size_t)255;
        return p;
    };
    float*    h1      = (float*)   alloc((size_t)BATCH * NC1 * 144 * sizeof(float));   // 2.95 MB
    float*    h2      = (float*)   alloc((size_t)BATCH * 800 * sizeof(float));         // 0.82 MB
    float*    bn_sums = (float*)   alloc(2 * NC2 * sizeof(float));
    float*    bn_ab   = (float*)   alloc(2 * NC2 * sizeof(float));
    _Float16* act     = (_Float16*)alloc((size_t)BATCH * FC1_IN * sizeof(_Float16));   // 0.41 MB
    _Float16* w1      = (_Float16*)alloc((size_t)FC1_PAD * FC1_IN * sizeof(_Float16)); // 0.82 MB
    float*    fc1o    = (float*)   alloc((size_t)BATCH * FC1_PAD * sizeof(float));     // 0.52 MB

    hipMemsetAsync(bn_sums, 0, 2 * NC2 * sizeof(float), stream);

    conv1_pool_kernel<<<(BATCH * NC1 * 144 + 255) / 256, 256, 0, stream>>>(
        x, conv1_w, conv1_b, h1);
    adder_pool_stats_kernel<<<BATCH, 256, 0, stream>>>(h1, adder_w, h2, bn_sums);
    bn_finalize_kernel<<<1, 64, 0, stream>>>(bn_sums, bn_g, bn_b, bn_ab);
    int convN = BATCH * FC1_IN + FC1_PAD * FC1_IN;
    to_f16_kernel<<<(convN + 255) / 256, 256, 0, stream>>>(h2, bn_ab, fc1_w, act, w1);
    fc1_wmma_kernel<<<64, 256, 0, stream>>>(act, w1, fc1_b, fc1o);  // 512 waves, 1 tile each
    fc2_lsm_kernel<<<BATCH, 32, 0, stream>>>(fc1o, fc2_w, fc2_b, (float*)d_out);
}
